// CylinderFeatureEncoder_65919158059356
// MI455X (gfx1250) — matrix operand order repaired and checked by reading, not compiled, not run on hardware
//
#include <hip/hip_runtime.h>
#include <hip/hip_fp16.h>

typedef _Float16 v16h __attribute__((ext_vector_type(16)));
typedef _Float16 v8h  __attribute__((ext_vector_type(8)));
typedef float    v8f  __attribute__((ext_vector_type(8)));
typedef float    v4f  __attribute__((ext_vector_type(4)));

#define WAVES   8
#define NBLK    1024
#define GRID    64
#define EPSBN   1e-5f

// ---- transposed-GEMM formulation:  H_out^T = W^T x H_in^T ----
// A operand  = W^T tile  (16 out-features x 32 in-features), lane = out-feature row
// B operand  = H_in^T    (32 in-features  x 16 points),      lane = point column
// C/D        = H_out^T   (16 out-features x 16 points),      lane = point, regs = 8 contiguous features

__device__ __forceinline__ v8f zero8() {
    v8f z;
#pragma unroll
    for (int i = 0; i < 8; ++i) z[i] = 0.f;
    return z;
}

__device__ __forceinline__ v16h cat8(v8h lo, v8h hi) {
    return __builtin_shufflevector(lo, hi, 0,1,2,3,4,5,6,7,8,9,10,11,12,13,14,15);
}

// A-matrix 16x32 f16 (ISA §7.12.2): lane L -> row n0+L%16; kb=(L<16)?0:8;
// elements 0..7 = K kb..kb+7, elements 8..15 = K kb+16..kb+23.  W^T stored [n][ldk], K contiguous.
__device__ __forceinline__ v16h ldW(const _Float16* wT, int ldk, int n0, int k0, int lane) {
    const _Float16* p = wT + (n0 + (lane & 15)) * ldk + k0 + ((lane & 16) ? 8 : 0);
    v8h lo = *(const v8h*)p;
    v8h hi = *(const v8h*)(p + 16);
    return cat8(lo, hi);
}

// B-matrix 32x16 f16: lane L -> point column L%16; kb=(L<16)?0:8;
// elements 0..7 = K kb..kb+7, elements 8..15 = K kb+16..kb+23.  Activations stored row-major [point][ldf].
__device__ __forceinline__ v16h ldAct(const _Float16* act, int ldf, int k0, int lane) {
    const _Float16* p = act + (lane & 15) * ldf + k0 + ((lane & 16) ? 8 : 0);
    v8h lo = *(const v8h*)p;
    v8h hi = *(const v8h*)(p + 16);
    return cat8(lo, hi);
}

// C/D 16x16 f32: lane = point L%16; reg r = feature n0 + mb + r, mb=(L>=16)?8:0.
// Bias + optional ReLU, pack to f16, single 16B store into [point][feature] LDS.
__device__ __forceinline__ void stAct(_Float16* act, int ldf, int n0, const float* bias,
                                      v8f acc, int lane, bool relu) {
    const int m  = lane & 15;
    const int nb = n0 + ((lane & 16) ? 8 : 0);
    v4f b0 = *(const v4f*)(bias + nb);
    v4f b1 = *(const v4f*)(bias + nb + 4);
    v8h hv;
#pragma unroll
    for (int r = 0; r < 8; ++r) {
        float v = acc[r] + ((r < 4) ? b0[r] : b1[r - 4]);
        if (relu) v = v > 0.f ? v : 0.f;
        hv[r] = (_Float16)v;
    }
    *(v8h*)(act + m * ldf + nb) = hv;
}

__device__ __forceinline__ v8f mm(v16h a, v16h b, v8f c) {
    return __builtin_amdgcn_wmma_f32_16x16x32_f16(false, a, false, b, (short)0, c, false, false);
}

__device__ __forceinline__ void wait_lds() {
    asm volatile("s_wait_dscnt 0" ::: "memory");
}

// ---- main fused kernel ----
// LDS: w0T[64][32] w1T[128][64] w2T[256][128] w3T[64][256] (f16) |
//      c0[64] c1[128] c2[256] c3[64] (f32) |
//      per-wave act: h1[16][64] h2[16][128] h3[16][256] (f16)
__global__ __launch_bounds__(256, 1)
void cyl_encoder_kernel(const float* __restrict__ pt_fea, const int* __restrict__ pt_ind,
                        const float* __restrict__ g0, const float* __restrict__ be0,
                        const float* __restrict__ m0, const float* __restrict__ vv0,
                        const float* __restrict__ g1, const float* __restrict__ be1,
                        const float* __restrict__ m1, const float* __restrict__ vv1,
                        const float* __restrict__ g2, const float* __restrict__ be2,
                        const float* __restrict__ m2, const float* __restrict__ vv2,
                        const float* __restrict__ g3, const float* __restrict__ be3,
                        const float* __restrict__ m3, const float* __restrict__ vv3,
                        const float* __restrict__ W0, const float* __restrict__ b0,
                        const float* __restrict__ W1, const float* __restrict__ b1,
                        const float* __restrict__ W2, const float* __restrict__ b2,
                        const float* __restrict__ W3, const float* __restrict__ b3,
                        float* __restrict__ out, int P, int Npts) {
    extern __shared__ char smem[];
    _Float16* w0T = (_Float16*)smem;            // 64*32   = 2048 h
    _Float16* w1T = w0T + 64 * 32;              // 128*64  = 8192 h
    _Float16* w2T = w1T + 128 * 64;             // 256*128 = 32768 h
    _Float16* w3T = w2T + 256 * 128;            // 64*256  = 16384 h
    float*    c0  = (float*)(w3T + 64 * 256);
    float*    c1  = c0 + 64;
    float*    c2  = c1 + 128;
    float*    c3  = c2 + 256;
    _Float16* actBase = (_Float16*)(c3 + 64);

    const int tid    = threadIdx.x;
    const int lane   = tid & 31;
    const int waveId = tid >> 5;

    // ---- fold BN into f16 W^T weights / f32 biases, once per persistent block ----
    for (int idx = tid; idx < 59392 + 512; idx += 256) {
        if (idx < 2048) {                                   // W0^T[n][k], k<3 live
            int n = idx >> 5, k = idx & 31;
            float val = 0.f;
            if (k < 3) {
                float s0 = g0[k] * rsqrtf(vv0[k] + EPSBN);
                float s1 = g1[n] * rsqrtf(vv1[n] + EPSBN);
                val = s0 * W0[k * 64 + n] * s1;
            }
            w0T[idx] = (_Float16)val;
        } else if (idx < 10240) {                           // W1^T
            int j = idx - 2048, n = j >> 6, k = j & 63;
            float s = g2[n] * rsqrtf(vv2[n] + EPSBN);
            w1T[j] = (_Float16)(W1[k * 128 + n] * s);
        } else if (idx < 43008) {                           // W2^T
            int j = idx - 10240, n = j >> 7, k = j & 127;
            float s = g3[n] * rsqrtf(vv3[n] + EPSBN);
            w2T[j] = (_Float16)(W2[k * 256 + n] * s);
        } else if (idx < 59392) {                           // W3^T (no BN after)
            int j = idx - 43008, n = j >> 8, k = j & 255;
            w3T[j] = (_Float16)W3[k * 64 + n];
        } else {                                            // folded biases
            int j = idx - 59392;
            if (j < 64) {
                int n = j;
                float s1 = g1[n] * rsqrtf(vv1[n] + EPSBN);
                float acc = b0[n] - m1[n];
#pragma unroll
                for (int k = 0; k < 3; ++k) {
                    float s0 = g0[k] * rsqrtf(vv0[k] + EPSBN);
                    float t0 = be0[k] - m0[k] * s0;
                    acc += t0 * W0[k * 64 + n];
                }
                c0[n] = acc * s1 + be1[n];
            } else if (j < 192) {
                int n = j - 64;
                float s = g2[n] * rsqrtf(vv2[n] + EPSBN);
                c1[n] = (b1[n] - m2[n]) * s + be2[n];
            } else if (j < 448) {
                int n = j - 192;
                float s = g3[n] * rsqrtf(vv3[n] + EPSBN);
                c2[n] = (b2[n] - m3[n]) * s + be3[n];
            } else {
                c3[j - 448] = b3[j - 448];
            }
        }
    }
    __syncthreads();

    _Float16* h1 = actBase + waveId * (16 * 448);
    _Float16* h2 = h1 + 16 * 64;
    _Float16* h3 = h2 + 16 * 128;

    const int NT     = P >> 4;                              // 16-point tiles
    const int stride = gridDim.x * WAVES;
    for (int tile = blockIdx.x * WAVES + waveId; tile < NT; tile += stride) {
        const int m = lane & 15;
        const int p = tile * 16 + m;

        // prefetch next tile's point data (hides the only global-load latency in the loop)
        {
            long pn = (long)(tile + stride) * 16 + m;
            if (pn < (long)P) {
                __builtin_prefetch(&pt_fea[3 * pn], 0, 1);
                __builtin_prefetch(&pt_ind[3 * pn], 0, 1);
            }
        }

        // per-lane segment id (lane handles point p for output scatter)
        const int ix = pt_ind[3 * p + 0], iy = pt_ind[3 * p + 1], iz = pt_ind[3 * p + 2];
        const int bb = p / Npts;
        const int sg = ((bb * GRID + iz) * GRID + iy) * GRID + ix;

        // B operand for layer 1 = X^T (3 live K rows); lanes>=16 (K=8..15,24..31) all zero pad
        v16h bx;
#pragma unroll
        for (int i = 0; i < 16; ++i) bx[i] = (_Float16)0.f;
        if (lane < 16) {
            bx[0] = (_Float16)pt_fea[3 * p + 0];
            bx[1] = (_Float16)pt_fea[3 * p + 1];
            bx[2] = (_Float16)pt_fea[3 * p + 2];
        }

        // ---- layer 1: W0^T[64x32] x X^T[32x16] ----
#pragma unroll
        for (int nt = 0; nt < 4; ++nt) {
            v8f acc = mm(ldW(w0T, 32, nt * 16, 0, lane), bx, zero8());
            stAct(h1, 64, nt * 16, c0, acc, lane, true);
        }
        wait_lds();

        // ---- layer 2: W1^T[128x64] x H1^T[64x16] ----
        v16h bf1[2];
#pragma unroll
        for (int kt = 0; kt < 2; ++kt) bf1[kt] = ldAct(h1, 64, kt * 32, lane);
#pragma unroll
        for (int nt = 0; nt < 8; ++nt) {
            v8f acc = zero8();
#pragma unroll
            for (int kt = 0; kt < 2; ++kt)
                acc = mm(ldW(w1T, 64, nt * 16, kt * 32, lane), bf1[kt], acc);
            stAct(h2, 128, nt * 16, c1, acc, lane, true);
        }
        wait_lds();

        // ---- layer 3: W2^T[256x128] x H2^T[128x16] ----
        v16h bf2[4];
#pragma unroll
        for (int kt = 0; kt < 4; ++kt) bf2[kt] = ldAct(h2, 128, kt * 32, lane);
#pragma unroll
        for (int nt = 0; nt < 16; ++nt) {
            v8f acc = zero8();
#pragma unroll
            for (int kt = 0; kt < 4; ++kt)
                acc = mm(ldW(w2T, 128, nt * 16, kt * 32, lane), bf2[kt], acc);
            stAct(h3, 256, nt * 16, c2, acc, lane, true);
        }
        wait_lds();

        // ---- layer 4: W3^T[64x256] x H3^T[256x16] + float atomic-max scatter ----
        v16h bf3[8];
#pragma unroll
        for (int kt = 0; kt < 8; ++kt) bf3[kt] = ldAct(h3, 256, kt * 32, lane);
        float* outp = out + (long)sg * 64;
#pragma unroll
        for (int nt = 0; nt < 4; ++nt) {
            v8f acc = zero8();
#pragma unroll
            for (int kt = 0; kt < 8; ++kt)
                acc = mm(ldW(w3T, 256, nt * 16, kt * 32, lane), bf3[kt], acc);
            const int nb = nt * 16 + ((lane & 16) ? 8 : 0);
            v4f bv0 = *(const v4f*)(c3 + nb);
            v4f bv1 = *(const v4f*)(c3 + nb + 4);
#pragma unroll
            for (int r = 0; r < 8; ++r) {
                float val = acc[r] + ((r < 4) ? bv0[r] : bv1[r - 4]);
                __hip_atomic_fetch_max(&outp[nb + r], val,
                                       __ATOMIC_RELAXED, __HIP_MEMORY_SCOPE_AGENT);
            }
        }
    }
}

// ---- init / finalize for segment_max semantics (empty voxels -> 0) ----
__global__ void init_out_kernel(float4* __restrict__ out, long n4) {
    const float ninf = -__builtin_inff();
    long i = (long)blockIdx.x * blockDim.x + threadIdx.x;
    const long st = (long)gridDim.x * blockDim.x;
    for (; i < n4; i += st) out[i] = make_float4(ninf, ninf, ninf, ninf);
}

__global__ void finalize_out_kernel(float4* __restrict__ out, long n4) {
    const float ninf = -__builtin_inff();
    long i = (long)blockIdx.x * blockDim.x + threadIdx.x;
    const long st = (long)gridDim.x * blockDim.x;
    for (; i < n4; i += st) {
        float4 v = out[i];
        v.x = (v.x == ninf) ? 0.f : v.x;
        v.y = (v.y == ninf) ? 0.f : v.y;
        v.z = (v.z == ninf) ? 0.f : v.z;
        v.w = (v.w == ninf) ? 0.f : v.w;
        out[i] = v;
    }
}

extern "C" void kernel_launch(void* const* d_in, const int* in_sizes, int n_in,
                              void* d_out, int out_size, void* d_ws, size_t ws_size,
                              hipStream_t stream) {
    (void)n_in; (void)d_ws; (void)ws_size;
    const float* pt_fea = (const float*)d_in[0];
    const int*   pt_ind = (const int*)d_in[1];
    const float* bn[16];
    for (int i = 0; i < 16; ++i) bn[i] = (const float*)d_in[2 + i];
    const float* W0 = (const float*)d_in[18]; const float* b0 = (const float*)d_in[19];
    const float* W1 = (const float*)d_in[20]; const float* b1 = (const float*)d_in[21];
    const float* W2 = (const float*)d_in[22]; const float* b2 = (const float*)d_in[23];
    const float* W3 = (const float*)d_in[24]; const float* b3 = (const float*)d_in[25];
    float* out = (float*)d_out;

    const int P    = in_sizes[0] / 3;   // total points (B*N)
    const int Npts = P / 2;             // points per batch (B=2)
    const long n4  = (long)out_size / 4;

    // weights 118784B + biases 2048B + act staging 8*14336B = 235520B
    const size_t ldsBytes = 59392 * 2 + 512 * 4 + WAVES * 16 * 448 * 2;
    (void)hipFuncSetAttribute((const void*)cyl_encoder_kernel,
                              hipFuncAttributeMaxDynamicSharedMemorySize, (int)ldsBytes);

    init_out_kernel<<<2048, 256, 0, stream>>>((float4*)out, n4);
    cyl_encoder_kernel<<<NBLK, 256, ldsBytes, stream>>>(
        pt_fea, pt_ind,
        bn[0], bn[1], bn[2], bn[3],  bn[4], bn[5], bn[6], bn[7],
        bn[8], bn[9], bn[10], bn[11], bn[12], bn[13], bn[14], bn[15],
        W0, b0, W1, b1, W2, b2, W3, b3,
        out, P, Npts);
    finalize_out_kernel<<<2048, 256, 0, stream>>>((float4*)out, n4);
}
